// MulAttontion_46239617908969
// MI455X (gfx1250) — compile-verified, matrix operand order checked by reading, not run on hardware
//
#include <hip/hip_runtime.h>
#include <hip/hip_bf16.h>
#include <math.h>

// ---------------------------------------------------------------------------
// MHA (B=32, C=256, H=W=32, heads=8, dh=32) for MI455X / gfx1250, wave32.
// All matmuls via v_wmma_f32_16x16x32_f16 (f16 in, f32 accumulate).
// Global->LDS staging uses GLOBAL_LOAD_ASYNC_TO_LDS_B128 (ASYNCcnt) with
// ping-pong double buffering; attention is flash-style (online softmax).
// Workspace ~81 MB:
//   XT[B][S][C], WQKV[768][256], WOc[256][256], BQKV[768] (f32),
//   QT[B][S][256], KT[B][S][256], V[B][256][S], OT[B][S][256]
// ---------------------------------------------------------------------------

#define B_    32
#define C_    256
#define S_    1024
#define HEADS_ 8
#define DH_   32
#define TOT_  256

typedef __attribute__((ext_vector_type(16))) _Float16 v16h;
typedef __attribute__((ext_vector_type(8)))  float    v8f;

union Frag16 {
    v16h v;
    _Float16 h[16];
    unsigned int w[8];
};

// K offset inside a 16-bit A/B fragment: VGPR j, lane-half 'half' holds the
// packed pair (K, K+1) with K = (j<4?0:16) + 2*(j&3) + 8*half  (ISA 7.12.2)
__device__ __forceinline__ int kbase(int j, int half) {
    return ((j & 4) << 2) + ((j & 3) << 1) + (half << 3);
}

__device__ __forceinline__ v8f wmma16(const Frag16& a, const Frag16& b, v8f c) {
    return __builtin_amdgcn_wmma_f32_16x16x32_f16(
        false, a.v, false, b.v, (short)0, c, false, false);
}

// Wave-relative LDS byte offset: generic pointers to __shared__ are
// SHARED_BASE(hi32) | lds_offset(lo32), and the async instructions add the
// wave's LDS_BASE themselves (ISA 10.x).
__device__ __forceinline__ unsigned lds_off(const void* p) {
    return (unsigned)(size_t)p;
}

// GLOBAL_LOAD_ASYNC_TO_LDS_B128: LDS[%0] = MEM[%1], tracked by ASYNCcnt.
__device__ __forceinline__ void async_copy_b128(unsigned lds_byte, const void* g) {
    asm volatile("global_load_async_to_lds_b128 %0, %1, off"
                 :: "v"(lds_byte), "v"(g) : "memory");
}
__device__ __forceinline__ void wait_async_le4() {
    asm volatile("s_wait_asynccnt 0x4" ::: "memory");
}
__device__ __forceinline__ void wait_async_0() {
    asm volatile("s_wait_asynccnt 0x0" ::: "memory");
}

// Stage one 64x32 A tile + 64x32 B tile (f16, K-contiguous rows) into LDS.
// 128 threads x 2 chunks x 16B per tile; 4 async instructions per thread.
__device__ __forceinline__ void stage_tiles_async(
    _Float16* lA, _Float16* lB,
    const _Float16* Abase, const _Float16* Bbase,
    int ldA, int ldB, int k0, int tid) {
    #pragma unroll
    for (int it = 0; it < 2; ++it) {
        const int ch = tid + it * 128;      // 256 chunks of 16B per 4KB tile
        const int row = ch >> 2, q = ch & 3;
        async_copy_b128(lds_off(lA + ch * 8), Abase + (size_t)row * ldA + k0 + q * 8);
        async_copy_b128(lds_off(lB + ch * 8), Bbase + (size_t)row * ldB + k0 + q * 8);
    }
}

// ---------------------------------------------------------------------------
// Conversions
// ---------------------------------------------------------------------------
__global__ void k_convert_x(const float* __restrict__ x,
                            _Float16* __restrict__ XT, int total) {
    int i = blockIdx.x * blockDim.x + threadIdx.x;
    if (i >= total) return;
    int c = i & (C_ - 1);
    int s = (i >> 8) & (S_ - 1);
    int b = i >> 18;
    XT[i] = (_Float16)x[((size_t)b * C_ + c) * S_ + s];
}

__global__ void k_convert_w(const float* __restrict__ Wq, const float* __restrict__ Wk,
                            const float* __restrict__ Wv, const float* __restrict__ bq,
                            const float* __restrict__ bk, const float* __restrict__ bv,
                            const float* __restrict__ Wo,
                            _Float16* __restrict__ WQKV, _Float16* __restrict__ WOc,
                            float* __restrict__ BQKV) {
    int i = blockIdx.x * blockDim.x + threadIdx.x;
    const int NW = 768 * 256;
    if (i < NW) {
        int rsel = i >> 16;           // 0: Wq, 1: Wk, 2: Wv (65536 each)
        const float* src = (rsel == 0) ? Wq : ((rsel == 1) ? Wk : Wv);
        WQKV[i] = (_Float16)src[i - rsel * 65536];
    } else if (i < NW + 65536) {
        WOc[i - NW] = (_Float16)Wo[i - NW];
    } else if (i < NW + 65536 + 768) {
        int j = i - NW - 65536;
        BQKV[j] = (j < 256) ? bq[j] : ((j < 512) ? bk[j - 256] : bv[j - 512]);
    }
}

// ---------------------------------------------------------------------------
// QKV projection: C[S=1024, N=768] = XT_b[S,256] * WQKV[768,256]^T  (NT GEMM)
// 128 threads (4 waves), tile 64x64, K-steps of 32, async ping-pong staging.
// ---------------------------------------------------------------------------
__global__ __launch_bounds__(128)
void k_gemm_qkv(const _Float16* __restrict__ XT, const _Float16* __restrict__ WQKV,
                const float* __restrict__ BQKV,
                _Float16* __restrict__ QT, _Float16* __restrict__ KT,
                _Float16* __restrict__ V) {
    __shared__ __align__(16) _Float16 lA[2][64 * 32];
    __shared__ __align__(16) _Float16 lB[2][64 * 32];
    const int b    = blockIdx.z;
    const int m0   = blockIdx.x * 64;
    const int n0   = blockIdx.y * 64;
    const int tid  = threadIdx.x;
    const int lane = tid & 31, wave = tid >> 5;
    const int half = lane >> 4, l16 = lane & 15;

    v8f acc[4] = {{}, {}, {}, {}};
    const _Float16* Abase = XT + ((size_t)b * S_ + m0) * C_;
    const _Float16* Bbase = WQKV + (size_t)n0 * C_;

    stage_tiles_async(lA[0], lB[0], Abase, Bbase, C_, C_, 0, tid);
    for (int ks = 0; ks < 8; ++ks) {
        const int buf = ks & 1;
        if (ks < 7) {
            stage_tiles_async(lA[buf ^ 1], lB[buf ^ 1], Abase, Bbase, C_, C_,
                              (ks + 1) * 32, tid);
            wait_async_le4();   // older batch (4 instrs) complete
        } else {
            wait_async_0();
        }
        __syncthreads();
        Frag16 a;
        #pragma unroll
        for (int j = 0; j < 8; ++j)
            a.w[j] = *(const unsigned int*)&lA[buf][(wave * 16 + l16) * 32 + kbase(j, half)];
        #pragma unroll
        for (int nf = 0; nf < 4; ++nf) {
            Frag16 bf;
            #pragma unroll
            for (int j = 0; j < 8; ++j)
                bf.w[j] = *(const unsigned int*)&lB[buf][(nf * 16 + l16) * 32 + kbase(j, half)];
            acc[nf] = wmma16(a, bf, acc[nf]);
        }
        __syncthreads();
    }

    #pragma unroll
    for (int nf = 0; nf < 4; ++nf) {
        const int n = n0 + nf * 16 + l16;
        const float bias = BQKV[n];
        #pragma unroll
        for (int r = 0; r < 8; ++r) {
            const int s = m0 + wave * 16 + r + half * 8;   // C layout: M=r+8*half
            const _Float16 hv = (_Float16)(acc[nf][r] + bias);
            if (n < 256)      QT[((size_t)b * S_ + s) * TOT_ + n]         = hv;
            else if (n < 512) KT[((size_t)b * S_ + s) * TOT_ + (n - 256)] = hv;
            else              V [((size_t)b * TOT_ + (n - 512)) * S_ + s] = hv;
        }
    }
}

// ---------------------------------------------------------------------------
// Flash attention: grid (S/64, HEADS, B), 128 threads (4 waves).
// K/V blocks staged once per block into LDS (async, ping-pong); each wave
// owns 16 query rows and loops over 16 key blocks of 64.
// ---------------------------------------------------------------------------
__global__ __launch_bounds__(128)
void k_flash(const _Float16* __restrict__ QT, const _Float16* __restrict__ KT,
             const _Float16* __restrict__ V, _Float16* __restrict__ OT) {
    __shared__ __align__(16) _Float16 lK[2][64 * 32];   // [t-local][d]
    __shared__ __align__(16) _Float16 lV[2][32 * 64];   // [d-local][t]
    __shared__ __align__(16) _Float16 lP[4][16 * 64];   // per-wave P tile
    const int b    = blockIdx.z, h = blockIdx.y;
    const int tid  = threadIdx.x;
    const int lane = tid & 31, wave = tid >> 5;
    const int half = lane >> 4, l16 = lane & 15;
    const int q0   = blockIdx.x * 64 + wave * 16;
    const float scale = 0.17677669529663687f;  // 1/sqrt(32)

    // Q fragment (A operand, K = DH = 32 -> a single WMMA K-step)
    Frag16 qf;
    #pragma unroll
    for (int j = 0; j < 8; ++j)
        qf.w[j] = *(const unsigned int*)(QT + ((size_t)b * S_ + (q0 + l16)) * TOT_
                                            + h * DH_ + kbase(j, half));

    float rowmax[8], rowsum[8];
    #pragma unroll
    for (int r = 0; r < 8; ++r) { rowmax[r] = -INFINITY; rowsum[r] = 0.f; }
    v8f oacc0 = {}, oacc1 = {};

    auto stage_kv = [&](int buf, int t0) {
        #pragma unroll
        for (int it = 0; it < 2; ++it) {
            const int ch = tid + it * 128;
            {   // K tile: 64 rows (t) x 32 (d)
                const int row = ch >> 2, part = ch & 3;
                async_copy_b128(lds_off(&lK[buf][ch * 8]),
                                KT + ((size_t)b * S_ + t0 + row) * TOT_ + h * DH_ + part * 8);
            }
            {   // V tile: 32 rows (d) x 64 (t)
                const int row = ch >> 3, part = ch & 7;
                async_copy_b128(lds_off(&lV[buf][ch * 8]),
                                V + ((size_t)b * TOT_ + h * DH_ + row) * S_ + t0 + part * 8);
            }
        }
    };

    stage_kv(0, 0);
    for (int kb = 0; kb < 16; ++kb) {
        const int buf = kb & 1;
        const int t0  = kb * 64;
        if (kb < 15) { stage_kv(buf ^ 1, t0 + 64); wait_async_le4(); }
        else         { wait_async_0(); }
        __syncthreads();   // K/V tile visible to all waves

        // scores tile [16 x 64] = 4 WMMAs
        v8f sc[4];
        #pragma unroll
        for (int nf = 0; nf < 4; ++nf) {
            Frag16 kf;
            #pragma unroll
            for (int j = 0; j < 8; ++j)
                kf.w[j] = *(const unsigned int*)&lK[buf][(nf * 16 + l16) * 32 + kbase(j, half)];
            v8f z = {};
            sc[nf] = wmma16(qf, kf, z);
        }

        // online softmax; row M = r + 8*half lives in lanes of this half
        #pragma unroll
        for (int r = 0; r < 8; ++r) {
            float m = fmaxf(fmaxf(sc[0][r], sc[1][r]), fmaxf(sc[2][r], sc[3][r]));
            #pragma unroll
            for (int off = 1; off < 16; off <<= 1)
                m = fmaxf(m, __shfl_xor(m, off, 32));
            const float nm    = fmaxf(rowmax[r], m);
            const float alpha = __expf(scale * (rowmax[r] - nm));
            rowmax[r] = nm;
            float psum = 0.f;
            #pragma unroll
            for (int nf = 0; nf < 4; ++nf) {
                const float p = __expf(scale * (sc[nf][r] - nm));
                sc[nf][r] = p;
                psum += p;
            }
            #pragma unroll
            for (int off = 1; off < 16; off <<= 1)
                psum += __shfl_xor(psum, off, 32);
            rowsum[r] = rowsum[r] * alpha + psum;
            oacc0[r]  = oacc0[r] * alpha;
            oacc1[r]  = oacc1[r] * alpha;
        }

        // C-layout P -> LDS -> A-layout (wave-private region; LDS ops from a
        // wave complete in order, so no block barrier is required here).
        #pragma unroll
        for (int nf = 0; nf < 4; ++nf)
            #pragma unroll
            for (int r = 0; r < 8; ++r)
                lP[wave][(r + half * 8) * 64 + nf * 16 + l16] = (_Float16)sc[nf][r];

        // O[16 x 32] += P[16 x 64] * V_h[64 x 32]   (K = 64 -> 2 steps)
        #pragma unroll
        for (int ks = 0; ks < 2; ++ks) {
            Frag16 pf;
            #pragma unroll
            for (int j = 0; j < 8; ++j)
                pf.w[j] = *(const unsigned int*)&lP[wave][l16 * 64 + ks * 32 + kbase(j, half)];
            #pragma unroll
            for (int nf = 0; nf < 2; ++nf) {
                Frag16 vf;
                #pragma unroll
                for (int j = 0; j < 8; ++j)
                    vf.w[j] = *(const unsigned int*)&lV[buf][(nf * 16 + l16) * 64
                                                            + ks * 32 + kbase(j, half)];
                if (nf == 0) oacc0 = wmma16(pf, vf, oacc0);
                else         oacc1 = wmma16(pf, vf, oacc1);
            }
        }
        __syncthreads();   // all reads of lK/lV[buf] done before it is re-staged
    }

    #pragma unroll
    for (int nf = 0; nf < 2; ++nf)
        #pragma unroll
        for (int r = 0; r < 8; ++r) {
            const int s = q0 + r + half * 8;
            const int d = h * DH_ + nf * 16 + l16;
            const float val = ((nf == 0) ? oacc0[r] : oacc1[r]) / rowsum[r];
            OT[((size_t)b * S_ + s) * TOT_ + d] = (_Float16)val;
        }
}

// ---------------------------------------------------------------------------
// Output projection + residuals:
// YT[S,256] = OT_b[S,256] * WOc[256,256]^T ; out[b,c,s] = 2*x + g*(YT[s,c]+bo[c])
// ---------------------------------------------------------------------------
__global__ __launch_bounds__(128)
void k_gemm_out(const _Float16* __restrict__ OT, const _Float16* __restrict__ WOc,
                const float* __restrict__ bo, const float* __restrict__ x,
                const float* __restrict__ gamma, float* __restrict__ out) {
    __shared__ __align__(16) _Float16 lA[2][64 * 32];
    __shared__ __align__(16) _Float16 lB[2][64 * 32];
    const int b    = blockIdx.z;
    const int m0   = blockIdx.x * 64;
    const int n0   = blockIdx.y * 64;
    const int tid  = threadIdx.x;
    const int lane = tid & 31, wave = tid >> 5;
    const int half = lane >> 4, l16 = lane & 15;

    v8f acc[4] = {{}, {}, {}, {}};
    const _Float16* Abase = OT + ((size_t)b * S_ + m0) * TOT_;
    const _Float16* Bbase = WOc + (size_t)n0 * TOT_;

    stage_tiles_async(lA[0], lB[0], Abase, Bbase, TOT_, TOT_, 0, tid);
    for (int ks = 0; ks < 8; ++ks) {
        const int buf = ks & 1;
        if (ks < 7) {
            stage_tiles_async(lA[buf ^ 1], lB[buf ^ 1], Abase, Bbase, TOT_, TOT_,
                              (ks + 1) * 32, tid);
            wait_async_le4();
        } else {
            wait_async_0();
        }
        __syncthreads();
        Frag16 a;
        #pragma unroll
        for (int j = 0; j < 8; ++j)
            a.w[j] = *(const unsigned int*)&lA[buf][(wave * 16 + l16) * 32 + kbase(j, half)];
        #pragma unroll
        for (int nf = 0; nf < 4; ++nf) {
            Frag16 bf;
            #pragma unroll
            for (int j = 0; j < 8; ++j)
                bf.w[j] = *(const unsigned int*)&lB[buf][(nf * 16 + l16) * 32 + kbase(j, half)];
            acc[nf] = wmma16(a, bf, acc[nf]);
        }
        __syncthreads();
    }

    const float g = gamma[0];
    #pragma unroll
    for (int nf = 0; nf < 4; ++nf) {
        const int c = n0 + nf * 16 + l16;
        const float bias = bo[c];
        #pragma unroll
        for (int r = 0; r < 8; ++r) {
            const int s = m0 + wave * 16 + r + half * 8;
            const size_t xi = ((size_t)b * C_ + c) * S_ + s;
            out[xi] = 2.f * x[xi] + g * (acc[nf][r] + bias);
        }
    }
}

// ---------------------------------------------------------------------------
extern "C" void kernel_launch(void* const* d_in, const int* in_sizes, int n_in,
                              void* d_out, int out_size, void* d_ws, size_t ws_size,
                              hipStream_t stream) {
    (void)in_sizes; (void)n_in; (void)out_size; (void)ws_size;
    const float* x     = (const float*)d_in[0];
    const float* Wq    = (const float*)d_in[1];
    const float* bq    = (const float*)d_in[2];
    const float* Wk    = (const float*)d_in[3];
    const float* bk    = (const float*)d_in[4];
    const float* Wv    = (const float*)d_in[5];
    const float* bv    = (const float*)d_in[6];
    const float* Wo    = (const float*)d_in[7];
    const float* bo    = (const float*)d_in[8];
    const float* gamma = (const float*)d_in[9];
    float* out = (float*)d_out;

    char* ws = (char*)d_ws;
    size_t off = 0;
    auto alloc = [&](size_t bytes) -> void* {
        void* p = ws + off;
        off += (bytes + 255) & ~(size_t)255;
        return p;
    };
    _Float16* XT   = (_Float16*)alloc((size_t)B_ * S_ * C_ * 2);   // 16 MB
    _Float16* WQKV = (_Float16*)alloc((size_t)768 * 256 * 2);
    _Float16* WOc  = (_Float16*)alloc((size_t)256 * 256 * 2);
    float*    BQKV = (float*)   alloc((size_t)768 * 4);
    _Float16* QT   = (_Float16*)alloc((size_t)B_ * S_ * TOT_ * 2); // 16 MB
    _Float16* KT   = (_Float16*)alloc((size_t)B_ * S_ * TOT_ * 2); // 16 MB
    _Float16* Vb   = (_Float16*)alloc((size_t)B_ * TOT_ * S_ * 2); // 16 MB
    _Float16* OT   = (_Float16*)alloc((size_t)B_ * S_ * TOT_ * 2); // 16 MB

    const int totalX = B_ * S_ * C_;                 // 8388608
    k_convert_x<<<(totalX + 255) / 256, 256, 0, stream>>>(x, XT, totalX);

    const int totalW = 768 * 256 + 256 * 256 + 768;  // 262912
    k_convert_w<<<(totalW + 255) / 256, 256, 0, stream>>>(Wq, Wk, Wv, bq, bk, bv, Wo,
                                                          WQKV, WOc, BQKV);

    k_gemm_qkv<<<dim3(S_ / 64, 768 / 64, B_), 128, 0, stream>>>(XT, WQKV, BQKV, QT, KT, Vb);
    k_flash  <<<dim3(S_ / 64, HEADS_, B_),    128, 0, stream>>>(QT, KT, Vb, OT);
    k_gemm_out<<<dim3(S_ / 64, C_ / 64, B_),  128, 0, stream>>>(OT, WOc, bo, x, gamma, out);
}
